// LSTM_61890478735972
// MI455X (gfx1250) — matrix operand order repaired and checked by reading, not compiled, not run on hardware
//
#include <hip/hip_runtime.h>
#include <hip/hip_bf16.h>
#include <stdint.h>

// ---------------------------------------------------------------------------
// LSTM forward for MI455X (gfx1250, wave32, WMMA bf16 -> f32 accumulate)
// T=512, B=64, I=1024, H=1024
// ---------------------------------------------------------------------------

#define T_STEPS 512
#define BATCH   64
#define IDIM    1024
#define HDIM    1024
#define GDIM    4096   // 4*H

typedef __attribute__((ext_vector_type(16))) __bf16        v16bf;
typedef __attribute__((ext_vector_type(8)))  float         v8f;
typedef __attribute__((ext_vector_type(4)))  unsigned int  u32x4;

union FragAB {            // 16 bf16 = 32 bytes = 8 VGPRs per lane
    v16bf bf;
    u32x4 q[2];
};

__device__ __forceinline__ uint16_t f2bf(float f) {
    unsigned u = __float_as_uint(f);
    unsigned r = u + 0x7FFFu + ((u >> 16) & 1u);   // round-to-nearest-even
    return (uint16_t)(r >> 16);
}

__device__ __forceinline__ float sigmoidf_(float x) {
    return 1.0f / (1.0f + __expf(-x));
}

// ---------------------------------------------------------------------------
// fp32 -> bf16 conversion (grid-stride)
// ---------------------------------------------------------------------------
__global__ void cvt_f32_bf16(const float* __restrict__ src,
                             uint16_t* __restrict__ dst, int n) {
    int i = blockIdx.x * blockDim.x + threadIdx.x;
    int stride = gridDim.x * blockDim.x;
    for (; i < n; i += stride) dst[i] = f2bf(src[i]);
}

// zero h0 (bf16) and c (fp32) state
__global__ void init_state(uint16_t* __restrict__ h0, float* __restrict__ c, int n) {
    int i = blockIdx.x * blockDim.x + threadIdx.x;
    if (i < n) { h0[i] = 0; c[i] = 0.0f; }
}

// ---------------------------------------------------------------------------
// x_gates[T*B, 4H] = input_bf16[T*B, I] @ w_ih_bf16[4H, I]^T + (b_ih + b_hh)
// block = 256 threads = 8 waves; wave tile 16x128 (8 N-tiles); block tile 64x256
// grid = (32768/64, 4096/256) = (512, 16)
// ---------------------------------------------------------------------------
__global__ __launch_bounds__(256) void gemm_xgates(
        const uint16_t* __restrict__ A,     // [32768, 1024] bf16
        const uint16_t* __restrict__ W,     // [4096, 1024] bf16
        const float* __restrict__ b_ih,
        const float* __restrict__ b_hh,
        float* __restrict__ xg)             // [32768, 4096] fp32
{
    const int tid  = threadIdx.x;
    const int lane = tid & 31;
    const int wv   = tid >> 5;
    const int mi   = wv & 3;          // M tile within block
    const int ng   = wv >> 2;         // N group (0/1)
    const int l15  = lane & 15;
    const int lhi  = lane >> 4;

    const int row_base = blockIdx.x * 64  + mi * 16;
    const int col_base = blockIdx.y * 256 + ng * 128;

    v8f acc[8] = {};
    float bias[8];
#pragma unroll
    for (int j = 0; j < 8; ++j) {
        int c = col_base + j * 16 + l15;
        bias[j] = b_ih[c] + b_hh[c];
    }

    const uint16_t* aptr = A + (size_t)(row_base + l15) * IDIM + lhi * 16;
    const uint16_t* bptr = W + (size_t)(col_base + l15) * IDIM + lhi * 16;

    for (int k0 = 0; k0 < IDIM; k0 += 32) {
        FragAB Af;
        Af.q[0] = *(const u32x4*)(aptr + k0);
        Af.q[1] = *(const u32x4*)(aptr + k0 + 8);
#pragma unroll
        for (int j = 0; j < 8; ++j) {
            FragAB Bf;
            const uint16_t* bp = bptr + (size_t)j * 16 * IDIM + k0;
            Bf.q[0] = *(const u32x4*)(bp);
            Bf.q[1] = *(const u32x4*)(bp + 8);
            acc[j] = __builtin_amdgcn_wmma_f32_16x16x32_bf16(
                false, Af.bf, false, Bf.bf, (short)0, acc[j], false, false);
        }
    }

#pragma unroll
    for (int j = 0; j < 8; ++j) {
        int col = col_base + j * 16 + l15;
#pragma unroll
        for (int r = 0; r < 8; ++r) {
            int row = row_base + r + lhi * 8;
            xg[(size_t)row * GDIM + col] = acc[j][r] + bias[j];
        }
    }
}

// ---------------------------------------------------------------------------
// One recurrent timestep:
//   gates = x_gates[t] + h_prev @ w_hh^T          (WMMA, result via LDS)
//   c = sigm(f)*c + sigm(i)*tanh(g); h = sigm(o)*tanh(c)
// Each block owns 32 hidden units (all 4 gate columns -> 128 GEMM cols) for
// all 64 batch rows.  grid = 1024/32 = 32 blocks, 256 threads (8 waves).
// Wave (mi, half): M tile mi (16 rows), one 16-col subtile per gate.
// ---------------------------------------------------------------------------
__global__ __launch_bounds__(256) void lstm_step(
        const float* __restrict__ xg,       // [32768, 4096] (includes biases)
        const uint16_t* __restrict__ whh,   // [4096, 1024] bf16
        const uint16_t* __restrict__ h_in,  // [64, 1024] bf16
        uint16_t* __restrict__ h_out,       // [64, 1024] bf16
        float* __restrict__ c,              // [64, 1024] fp32 state
        float* __restrict__ out,            // [T, 64, 1024] fp32
        float* __restrict__ hT,             // [64, 1024]
        float* __restrict__ cT,             // [64, 1024]
        int t)
{
    __shared__ float lds[64 * 132];         // padded stride: bank-conflict free

    const int tid  = threadIdx.x;
    const int lane = tid & 31;
    const int wv   = tid >> 5;
    const int mi   = wv & 3;
    const int half = wv >> 2;
    const int l15  = lane & 15;
    const int lhi  = lane >> 4;
    const int n0   = blockIdx.x * 32;       // hidden-unit chunk base

    v8f acc[4] = {};

    const uint16_t* aptr  = h_in + (size_t)(mi * 16 + l15) * HDIM + lhi * 16;
    const uint16_t* bbase = whh + (size_t)(n0 + half * 16 + l15) * HDIM + lhi * 16;

    for (int k0 = 0; k0 < HDIM; k0 += 32) {
        FragAB Af;
        Af.q[0] = *(const u32x4*)(aptr + k0);
        Af.q[1] = *(const u32x4*)(aptr + k0 + 8);
#pragma unroll
        for (int g = 0; g < 4; ++g) {
            FragAB Bf;
            const uint16_t* bp = bbase + (size_t)g * HDIM * HDIM + k0;
            Bf.q[0] = *(const u32x4*)(bp);
            Bf.q[1] = *(const u32x4*)(bp + 8);
            acc[g] = __builtin_amdgcn_wmma_f32_16x16x32_bf16(
                false, Af.bf, false, Bf.bf, (short)0, acc[g], false, false);
        }
    }

    // scatter recurrent GEMM result to LDS: row = batch, col = g*32 + local n
#pragma unroll
    for (int g = 0; g < 4; ++g) {
        int colLocal = g * 32 + half * 16 + l15;
#pragma unroll
        for (int r = 0; r < 8; ++r) {
            int m = mi * 16 + r + lhi * 8;
            lds[m * 132 + colLocal] = acc[g][r];
        }
    }
    __syncthreads();

    // elementwise LSTM update: 64 rows x 32 units; 4 threads/row, 8 units each
    const int b      = tid >> 2;
    const int nlbase = (tid & 3) * 8;
    const float* xrow = xg + ((size_t)t * BATCH + b) * GDIM + n0;
    float* crow = c + b * HDIM + n0;
    float* orow = out + ((size_t)t * BATCH + b) * HDIM + n0;
    uint16_t* hrow = h_out + b * HDIM + n0;

#pragma unroll
    for (int e = 0; e < 8; ++e) {
        int nl = nlbase + e;
        float iv = lds[b * 132 + nl]      + xrow[nl];
        float fv = lds[b * 132 + 32 + nl] + xrow[HDIM + nl];
        float gv = lds[b * 132 + 64 + nl] + xrow[2 * HDIM + nl];
        float ov = lds[b * 132 + 96 + nl] + xrow[3 * HDIM + nl];

        float is = sigmoidf_(iv);
        float fs = sigmoidf_(fv);
        float gs = tanhf(gv);
        float os = sigmoidf_(ov);

        float cn = fs * crow[nl] + is * gs;
        crow[nl] = cn;
        float h = os * tanhf(cn);
        orow[nl] = h;
        hrow[nl] = f2bf(h);

        if (t == T_STEPS - 1) {
            hT[b * HDIM + n0 + nl] = h;
            cT[b * HDIM + n0 + nl] = cn;
        }
    }
}

// ---------------------------------------------------------------------------
extern "C" void kernel_launch(void* const* d_in, const int* in_sizes, int n_in,
                              void* d_out, int out_size, void* d_ws, size_t ws_size,
                              hipStream_t stream) {
    const float* input = (const float*)d_in[0];   // [T, B, I]
    const float* w_ih  = (const float*)d_in[1];   // [4H, I]
    const float* w_hh  = (const float*)d_in[2];   // [4H, H]
    const float* b_ih  = (const float*)d_in[3];   // [4H]
    const float* b_hh  = (const float*)d_in[4];   // [4H]
    float* out = (float*)d_out;
    float* hT  = out + (size_t)T_STEPS * BATCH * HDIM;
    float* cT  = hT + (size_t)BATCH * HDIM;

    // Workspace layout
    char* ws = (char*)d_ws;
    size_t off = 0;
    float*    xg     = (float*)(ws + off);    off += (size_t)T_STEPS * BATCH * GDIM * 4; // 512 MB
    uint16_t* in_bf  = (uint16_t*)(ws + off); off += (size_t)T_STEPS * BATCH * IDIM * 2; //  64 MB
    uint16_t* wih_bf = (uint16_t*)(ws + off); off += (size_t)GDIM * IDIM * 2;            //   8 MB
    uint16_t* whh_bf = (uint16_t*)(ws + off); off += (size_t)GDIM * HDIM * 2;            //   8 MB
    uint16_t* hbuf   = (uint16_t*)(ws + off); off += (size_t)2 * BATCH * HDIM * 2;       // 256 KB
    float*    cbuf   = (float*)(ws + off);    off += (size_t)BATCH * HDIM * 4;           // 256 KB

    const int nIn  = T_STEPS * BATCH * IDIM;   // 33,554,432
    const int nW   = GDIM * IDIM;              //  4,194,304
    const int nSt  = BATCH * HDIM;             //     65,536

    // 1) convert to bf16 + zero-init state
    cvt_f32_bf16<<<4096, 256, 0, stream>>>(input, in_bf, nIn);
    cvt_f32_bf16<<<2048, 256, 0, stream>>>(w_ih, wih_bf, nW);
    cvt_f32_bf16<<<2048, 256, 0, stream>>>(w_hh, whh_bf, nW);
    init_state<<<nSt / 256, 256, 0, stream>>>(hbuf, cbuf, nSt);

    // 2) big input-projection GEMM (bias fused)
    gemm_xgates<<<dim3(T_STEPS * BATCH / 64, GDIM / 256), 256, 0, stream>>>(
        in_bf, wih_bf, b_ih, b_hh, xg);

    // 3) sequential recurrent scan: one kernel per timestep
    for (int t = 0; t < T_STEPS; ++t) {
        const uint16_t* h_in = hbuf + (size_t)(t & 1) * nSt;
        uint16_t* h_out      = hbuf + (size_t)((t + 1) & 1) * nSt;
        lstm_step<<<HDIM / 32, 256, 0, stream>>>(
            xg, whh_bf, h_in, h_out, cbuf, out, hT, cT, t);
    }
}
